// MaskedCrossAttention_57346403336697
// MI455X (gfx1250) — compile-verified, matrix operand order checked by reading, not run on hardware
//
#include <hip/hip_runtime.h>
#include <hip/hip_bf16.h>

// ---------------------------------------------------------------------------
// MaskedCrossAttention for MI455X (gfx1250, wave32, WMMA + async-LDS copies)
//   B=32 T=256 V=1000 C=1024 H=8 DH=64 INNER=512 K=5
//
// Pipeline:
//   1) convert vision -> bf16 ; transpose+convert Wq/Wkv/Wout -> bf16 [N][K]
//   2) mask -> first-K ascending indices (== -top_k(-idxs,K))
//   3) LayerNorm(x) -> bf16
//   4) Q  = xn  @ Wq  * SCALE   (WMMA bf16, f32 out)
//   5) KV = vision @ Wkv        (WMMA bf16, bf16 out; dense beats gather-GEMM)
//   6) 5-wide masked softmax attention (one wave per (b,t,head))
//   7) Y  = attn_out @ Wout -> d_out (WMMA bf16, f32 out)
//
// GEMM inner loop: double-buffered LDS fed by global_load_async_to_lds_b128
// (ASYNCcnt), fragments via ds_load_b128, 8x v_wmma_f32_16x16x32_bf16.
// ---------------------------------------------------------------------------

#define BB 32
#define TT 256
#define VV 1000
#define CC 1024
#define HH 8
#define DH 64
#define INNER 512
#define KK 5
#define SCALE 0.125f

typedef float v8f __attribute__((ext_vector_type(8)));
typedef __bf16 v16bf __attribute__((ext_vector_type(16)));
typedef __bf16 bf16x8 __attribute__((ext_vector_type(8)));

static __device__ __forceinline__ unsigned short f2bf_bits(float f) {
  union { float f; unsigned u; } v; v.f = f;
  unsigned r = v.u + 0x7FFFu + ((v.u >> 16) & 1u);   // RNE
  return (unsigned short)(r >> 16);
}
static __device__ __forceinline__ float bf2f_bits(unsigned short h) {
  union { unsigned u; float f; } v; v.u = ((unsigned)h) << 16; return v.f;
}

// ---------------------------------------------------------------------------
// Kernel 0a: elementwise f32 -> bf16 (vision), grid-stride, float4 granules.
// ---------------------------------------------------------------------------
__global__ __launch_bounds__(256) void mca_convert_bf16(
    const float* __restrict__ src, unsigned short* __restrict__ dst, int n4) {
  for (size_t i = (size_t)blockIdx.x * blockDim.x + threadIdx.x; i < (size_t)n4;
       i += (size_t)gridDim.x * blockDim.x) {
    const float4 d = ((const float4*)src)[i];
    uint2 o;
    o.x = (unsigned)f2bf_bits(d.x) | ((unsigned)f2bf_bits(d.y) << 16);
    o.y = (unsigned)f2bf_bits(d.z) | ((unsigned)f2bf_bits(d.w) << 16);
    ((uint2*)dst)[i] = o;
  }
}

// ---------------------------------------------------------------------------
// Kernel 0b: transpose + convert  W[K][N] f32  ->  Wt[N][K] bf16  (32x32 tiles)
// ---------------------------------------------------------------------------
__global__ __launch_bounds__(256) void mca_transpose_bf16(
    const float* __restrict__ src, unsigned short* __restrict__ dst,
    int Kd, int N) {
  __shared__ float tile[32][33];
  const int n0 = blockIdx.x * 32, k0 = blockIdx.y * 32;
  const int tx = threadIdx.x & 31, ty = threadIdx.x >> 5;   // ty: 0..7
#pragma unroll
  for (int i = 0; i < 4; ++i)
    tile[ty + i * 8][tx] = src[(size_t)(k0 + ty + i * 8) * N + n0 + tx];
  __syncthreads();
#pragma unroll
  for (int i = 0; i < 4; ++i)
    dst[(size_t)(n0 + ty + i * 8) * Kd + k0 + tx] = f2bf_bits(tile[tx][ty + i * 8]);
}

// ---------------------------------------------------------------------------
// Kernel 1: per (b,t) ascending scan of mask column -> first K valid v's,
// padded with V (== reference -top_k(-idxs, K)).
// ---------------------------------------------------------------------------
__global__ __launch_bounds__(256) void mca_topk_kernel(
    const int* __restrict__ mask, int* __restrict__ idxout) {
  const int b = blockIdx.x;
  const int t = threadIdx.x;
  int out[KK];
#pragma unroll
  for (int k = 0; k < KK; ++k) out[k] = VV;
  int cnt = 0;
  const size_t base = (size_t)b * VV * TT + t;
  for (int v = 0; v < VV; ++v) {
    int m = mask[base + (size_t)v * TT];
    if (m != 0 && cnt < KK) { out[cnt] = v; ++cnt; }
  }
  const size_t o = ((size_t)(b * TT + t)) * KK;
#pragma unroll
  for (int k = 0; k < KK; ++k) idxout[o + k] = out[k];
}

// ---------------------------------------------------------------------------
// Kernel 2: LayerNorm over C=1024, one block per (b,t) row; bf16 output.
// ---------------------------------------------------------------------------
__global__ __launch_bounds__(256) void mca_ln_kernel(
    const float* __restrict__ x, const float* __restrict__ w,
    const float* __restrict__ bias, unsigned short* __restrict__ xn) {
  __shared__ float red[256];
  const int tid = threadIdx.x;
  const size_t base = (size_t)blockIdx.x * CC;
  float lv[4];
  float s = 0.f;
#pragma unroll
  for (int j = 0; j < 4; ++j) { lv[j] = x[base + tid + j * 256]; s += lv[j]; }
  red[tid] = s; __syncthreads();
  for (int off = 128; off > 0; off >>= 1) {
    if (tid < off) red[tid] += red[tid + off];
    __syncthreads();
  }
  const float mu = red[0] * (1.f / (float)CC);
  __syncthreads();
  s = 0.f;
#pragma unroll
  for (int j = 0; j < 4; ++j) { float d = lv[j] - mu; s += d * d; }
  red[tid] = s; __syncthreads();
  for (int off = 128; off > 0; off >>= 1) {
    if (tid < off) red[tid] += red[tid + off];
    __syncthreads();
  }
  const float rs = rsqrtf(red[0] * (1.f / (float)CC) + 1e-5f);
#pragma unroll
  for (int j = 0; j < 4; ++j) {
    const int i = tid + j * 256;
    xn[base + i] = f2bf_bits((lv[j] - mu) * rs * w[i] + bias[i]);
  }
}

// ---------------------------------------------------------------------------
// Kernel 3: bf16 WMMA GEMM  C[M,N] = alpha * A[M,K] * Bt[N,K]^T
//   A row-major bf16, Bt row-major bf16 (pre-transposed weights).
//   block = 256 threads (8 waves), tile 128x128, K-step 32, double-buffered
//   LDS fed by global_load_async_to_lds_b128 (ASYNCcnt).
//   M%128==0, N%128==0, K%32==0 (holds for all three calls).
// ---------------------------------------------------------------------------
#define LDT 40  // padded LDS row stride (bf16); rows 80B => 16B-aligned chunks

// one 128x32 bf16 panel = 512 x 16B chunks; 256 threads -> 2 async copies each
#define STAGE_ASYNC(LDSBUF, GBASE, R0, KD, KKOFF)                               \
  _Pragma("unroll")                                                             \
  for (int r_ = 0; r_ < 2; ++r_) {                                              \
    const int e_ = tid + r_ * 256;                                              \
    const int row_ = e_ >> 2, c8_ = e_ & 3;                                     \
    const unsigned short* gp_ =                                                 \
        (GBASE) + (size_t)((R0) + row_) * (KD) + (KKOFF) + c8_ * 8;             \
    const unsigned lo_ = (unsigned)(size_t)(&(LDSBUF)[row_ * LDT + c8_ * 8]);   \
    asm volatile("global_load_async_to_lds_b128 %0, %1, off"                    \
                 :: "v"(lo_), "v"((unsigned long long)(size_t)gp_) : "memory"); \
  }

template <int OUT_BF16>
__global__ __launch_bounds__(256) void mca_gemm_wmma(
    const unsigned short* __restrict__ Aptr,   // [M][Kd] bf16
    const unsigned short* __restrict__ Btp,    // [N][Kd] bf16
    void* __restrict__ Cptr, int M, int N, int Kd, float alpha) {
  __shared__ __bf16 As[2][128 * LDT];
  __shared__ __bf16 Bs[2][128 * LDT];

  const int tid  = threadIdx.x;
  const int m0   = blockIdx.y * 128;
  const int n0   = blockIdx.x * 128;
  const int wave = tid >> 5;
  const int lane = tid & 31;
  const int wm   = wave & 3;          // wave M group (0..3) -> 32 rows
  const int wn   = wave >> 2;         // wave N group (0..1) -> 64 cols
  const int mrow = lane & 15;
  const int half = lane >> 4;

  v8f acc[2][4] = {};

  const int nt = Kd >> 5;
  // prologue: stage tile 0 into buffer 0 (4 async b128 per thread)
  STAGE_ASYNC(As[0], Aptr, m0, Kd, 0)
  STAGE_ASYNC(Bs[0], Btp, n0, Kd, 0)

  for (int it = 0; it < nt; ++it) {
    const int cur = it & 1;
    const bool more = (it + 1) < nt;
    if (more) {                       // stage tile it+1 into other buffer
      const int kk = (it + 1) << 5;
      STAGE_ASYNC(As[cur ^ 1], Aptr, m0, Kd, kk)
      STAGE_ASYNC(Bs[cur ^ 1], Btp, n0, Kd, kk)
      asm volatile("s_wait_asynccnt 0x4" ::: "memory");  // tile it complete
    } else {
      asm volatile("s_wait_asynccnt 0x0" ::: "memory");
    }
    __syncthreads();                  // tile it visible to all waves

    // ---- fragments (two 16B ds reads each) + 8 WMMAs ----
    v16bf afrag[2], bfrag[4];
#pragma unroll
    for (int i = 0; i < 2; ++i) {
      const int base = (wm * 32 + i * 16 + mrow) * LDT;
      bf16x8 lo = *(const bf16x8*)(&As[cur][base + half * 8]);
      bf16x8 hi = *(const bf16x8*)(&As[cur][base + 16 + half * 8]);
#pragma unroll
      for (int e = 0; e < 8; ++e) { afrag[i][e] = lo[e]; afrag[i][8 + e] = hi[e]; }
    }
#pragma unroll
    for (int j = 0; j < 4; ++j) {
      const int base = (wn * 64 + j * 16 + mrow) * LDT;
      bf16x8 lo = *(const bf16x8*)(&Bs[cur][base + half * 8]);
      bf16x8 hi = *(const bf16x8*)(&Bs[cur][base + 16 + half * 8]);
#pragma unroll
      for (int e = 0; e < 8; ++e) { bfrag[j][e] = lo[e]; bfrag[j][8 + e] = hi[e]; }
    }
#pragma unroll
    for (int i = 0; i < 2; ++i)
#pragma unroll
      for (int j = 0; j < 4; ++j)
        acc[i][j] = __builtin_amdgcn_wmma_f32_16x16x32_bf16(
            false, afrag[i], false, bfrag[j], (short)0, acc[i][j], false, false);
    __syncthreads();                  // all waves done reading buf `cur`
  }

  // ---- epilogue: C/D layout lane(L%16)=N, vgpr r -> M = r + 8*(L/16) ----
#pragma unroll
  for (int i = 0; i < 2; ++i)
#pragma unroll
    for (int j = 0; j < 4; ++j)
#pragma unroll
      for (int r = 0; r < 8; ++r) {
        const int m = m0 + wm * 32 + i * 16 + half * 8 + r;
        const int n = n0 + wn * 64 + j * 16 + mrow;
        const float val = acc[i][j][r] * alpha;
        if (OUT_BF16)
          ((unsigned short*)Cptr)[(size_t)m * N + n] = f2bf_bits(val);
        else
          ((float*)Cptr)[(size_t)m * N + n] = val;
      }
}

// ---------------------------------------------------------------------------
// Kernel 4: attention core. One block per (b,t), one wave per head.
// Each lane owns 2 of DH=64 dims; dot products wave-reduced via shfl_xor.
// ---------------------------------------------------------------------------
__global__ __launch_bounds__(256) void mca_attn_kernel(
    const float* __restrict__ q, const unsigned short* __restrict__ kv,
    const int* __restrict__ idx, unsigned short* __restrict__ aout) {
  const int bt = blockIdx.x;
  const int b = bt / TT;
  const int h = threadIdx.x >> 5;     // wave id == head
  const int lane = threadIdx.x & 31;
  const int d0 = lane * 2;

  const size_t qo = (size_t)bt * INNER + h * DH + d0;
  const float q0 = q[qo];             // SCALE already folded into Q GEMM
  const float q1 = q[qo + 1];

  int   ids[KK];
  float sim[KK];
#pragma unroll
  for (int k = 0; k < KK; ++k) {
    const int id = idx[(size_t)bt * KK + k];
    ids[k] = id;
    float dot = 0.f;
    if (id < VV) {
      const size_t base = ((size_t)b * VV + id) * (2 * INNER) + h * DH + d0;
      const unsigned pk = *(const unsigned*)(&kv[base]);
      dot = q0 * bf2f_bits((unsigned short)(pk & 0xFFFF)) +
            q1 * bf2f_bits((unsigned short)(pk >> 16));
    }
#pragma unroll
    for (int off = 16; off > 0; off >>= 1) dot += __shfl_xor(dot, off, 32);
    sim[k] = (id < VV) ? dot : -10000.0f;   // invalid: 0*W + (-1e4)
  }

  float mx = sim[0];
#pragma unroll
  for (int k = 1; k < KK; ++k) mx = fmaxf(mx, sim[k]);
  float ssum = 0.f;
  float p[KK];
#pragma unroll
  for (int k = 0; k < KK; ++k) { p[k] = __expf(sim[k] - mx); ssum += p[k]; }
  const float inv = 1.f / ssum;

  float o0 = 0.f, o1 = 0.f;
#pragma unroll
  for (int k = 0; k < KK; ++k) {
    if (ids[k] < VV) {
      const float a = p[k] * inv;
      const size_t base = ((size_t)b * VV + ids[k]) * (2 * INNER) + INNER + h * DH + d0;
      const unsigned pv = *(const unsigned*)(&kv[base]);
      o0 += a * bf2f_bits((unsigned short)(pv & 0xFFFF));
      o1 += a * bf2f_bits((unsigned short)(pv >> 16));
    }
  }
  const unsigned out = (unsigned)f2bf_bits(o0) | ((unsigned)f2bf_bits(o1) << 16);
  *(unsigned*)(&aout[(size_t)bt * INNER + h * DH + d0]) = out;
}

// ---------------------------------------------------------------------------
// Host launcher
// ---------------------------------------------------------------------------
extern "C" void kernel_launch(void* const* d_in, const int* in_sizes, int n_in,
                              void* d_out, int out_size, void* d_ws, size_t ws_size,
                              hipStream_t stream) {
  const float* x      = (const float*)d_in[0];
  const float* vision = (const float*)d_in[1];
  const float* ln_w   = (const float*)d_in[2];
  const float* ln_b   = (const float*)d_in[3];
  const float* Wq     = (const float*)d_in[4];
  const float* Wkv    = (const float*)d_in[5];
  const float* Wout   = (const float*)d_in[6];
  const int*   mask   = (const int*)d_in[7];
  (void)in_sizes; (void)n_in; (void)out_size; (void)ws_size;

  // Workspace carve-up (bytes, all 16B-aligned)
  char* ws = (char*)d_ws;
  size_t off = 0;
  unsigned short* xn   = (unsigned short*)(ws + off); off += (size_t)BB*TT*CC*2;        // 16 MB
  float*          qbuf = (float*)(ws + off);          off += (size_t)BB*TT*INNER*4;     // 16 MB
  unsigned short* kvb  = (unsigned short*)(ws + off); off += (size_t)BB*VV*2*INNER*2;   // 62.5 MB
  unsigned short* aout = (unsigned short*)(ws + off); off += (size_t)BB*TT*INNER*2;     // 8 MB
  unsigned short* visb = (unsigned short*)(ws + off); off += (size_t)BB*VV*CC*2;        // 62.5 MB
  unsigned short* WqT  = (unsigned short*)(ws + off); off += (size_t)INNER*CC*2;        // 1 MB
  unsigned short* WkvT = (unsigned short*)(ws + off); off += (size_t)(2*INNER)*CC*2;    // 2 MB
  unsigned short* WoutT= (unsigned short*)(ws + off); off += (size_t)CC*INNER*2;        // 1 MB
  int*            idxb = (int*)(ws + off);

  // 0) precision conversions (off critical path of GEMM inner loops)
  mca_convert_bf16<<<4096, 256, 0, stream>>>(vision, visb, (BB * VV * CC) / 4);
  { dim3 g(INNER / 32, CC / 32);       mca_transpose_bf16<<<g, 256, 0, stream>>>(Wq,   WqT,   CC,    INNER); }
  { dim3 g((2 * INNER) / 32, CC / 32); mca_transpose_bf16<<<g, 256, 0, stream>>>(Wkv,  WkvT,  CC,    2 * INNER); }
  { dim3 g(CC / 32, INNER / 32);       mca_transpose_bf16<<<g, 256, 0, stream>>>(Wout, WoutT, INNER, CC); }

  // 1) mask -> per-(b,t) sorted valid indices, padded with V
  mca_topk_kernel<<<BB, TT, 0, stream>>>(mask, idxb);

  // 2) LayerNorm -> bf16
  mca_ln_kernel<<<BB * TT, 256, 0, stream>>>(x, ln_w, ln_b, xn);

  // 3) Q = (xn @ Wq) * SCALE : M=8192 N=512 K=1024, f32 out
  {
    dim3 grid(INNER / 128, (BB * TT) / 128);
    mca_gemm_wmma<0><<<grid, 256, 0, stream>>>(xn, WqT, (void*)qbuf,
                                               BB * TT, INNER, CC, SCALE);
  }
  // 4) KV = vision @ Wkv : M=32000 N=1024 K=1024, bf16 out
  {
    dim3 grid((2 * INNER) / 128, (BB * VV) / 128);
    mca_gemm_wmma<1><<<grid, 256, 0, stream>>>(visb, WkvT, (void*)kvb,
                                               BB * VV, 2 * INNER, CC, 1.0f);
  }
  // 5) attention core -> aout (bf16)
  mca_attn_kernel<<<BB * TT, 256, 0, stream>>>(qbuf, kvb, idxb, aout);

  // 6) Y = aout @ Wout : M=8192 N=1024 K=512, f32 out -> d_out
  {
    dim3 grid(CC / 128, (BB * TT) / 128);
    mca_gemm_wmma<0><<<grid, 256, 0, stream>>>(aout, WoutT, d_out,
                                               BB * TT, CC, INNER, 1.0f);
  }
}
